// MiniMHCLM_62466004353588
// MI455X (gfx1250) — matrix operand order repaired and checked by reading, not compiled, not run on hardware
//
#include <hip/hip_runtime.h>

#define HC        4
#define C_DIM     256
#define K_DIM     1024      // HC*C
#define M_COEF    24        // HC*HC + 2*HC
#define TMAX_IT   8
#define VOCAB     50257
#define NTOK      4096      // B*S
#define RMS_EPS   1e-6f
#define PRE_EPS   1e-4f
#define SINK_EPS  1e-6f
#define POST_MULT 2.0f

typedef __attribute__((ext_vector_type(16))) __bf16 v16bf;
typedef __attribute__((ext_vector_type(8)))  float  v8f;
typedef __attribute__((ext_vector_type(4)))  unsigned int u32x4;
typedef __attribute__((ext_vector_type(8)))  int          i32x8;
typedef __attribute__((ext_vector_type(4)))  int          i32x4;

__device__ inline float bf16_to_f32(unsigned short h) {
  union { unsigned int u; float f; } x;
  x.u = ((unsigned int)h) << 16;
  return x.f;
}
// branch-free round-to-nearest-even f32 -> bf16 (finite inputs)
__device__ inline unsigned short f32_to_bf16(float f) {
  union { float f; unsigned int u; } x;
  x.f = f;
  return (unsigned short)((x.u + 0x7FFFu + ((x.u >> 16) & 1u)) >> 16);
}
// pack two f32 -> bf16x2: 2x (bfe+add3) + 1x v_perm_b32
__device__ inline unsigned pack_bf16x2(float a, float b) {
#if __has_builtin(__builtin_amdgcn_cvt_pk_bf16_f32)
  typedef __attribute__((ext_vector_type(2))) __bf16 v2bf;
  union { v2bf v; unsigned u; } r;
  r.v = __builtin_amdgcn_cvt_pk_bf16_f32(a, b);
  return r.u;
#else
  union { float f; unsigned int u; } x, y;
  x.f = a; y.f = b;
  const unsigned ra = x.u + 0x7FFFu + ((x.u >> 16) & 1u);
  const unsigned rb = y.u + 0x7FFFu + ((y.u >> 16) & 1u);
  return __builtin_amdgcn_perm(rb, ra, 0x07060302u);  // {rb[31:16], ra[31:16]}
#endif
}
__device__ inline float sigmoidf(float v) { return 1.0f / (1.0f + __expf(-v)); }

// -------------------- Kernel 1: per-token MHC pre-pass --------------------
__global__ __launch_bounds__(256) void mhc_token_kernel(
    const int*            __restrict__ ids,
    const float*          __restrict__ embed,     // [VOCAB][1024]
    const float*          __restrict__ w_inner,   // [256][256]
    const unsigned short* __restrict__ phi,       // [1024][24] bf16
    const float*          __restrict__ bvec,      // [24]
    const float*          __restrict__ a_pre,
    const float*          __restrict__ a_post,
    const float*          __restrict__ a_res,
    unsigned short*       __restrict__ xm)        // [NTOK][1024] bf16
{
  __shared__ float xf[K_DIM];
  __shared__ float part[8 * M_COEF];
  __shared__ float logits[M_COEF];
  __shared__ float hpre[HC], hpost[HC], hres[HC * HC];
  __shared__ float xin[C_DIM];
  __shared__ float ssum, srms;

  const int tid = threadIdx.x;
  const int tok = blockIdx.x;
  const long base = (long)ids[tok] * K_DIM;

  if (tid == 0) ssum = 0.0f;
  __syncthreads();

  float lsum = 0.0f;
#pragma unroll
  for (int i = 0; i < 4; ++i) {
    const int k = tid * 4 + i;
    const float v = bf16_to_f32(f32_to_bf16(embed[base + k]));
    xf[k] = v;
    lsum += v * v;
  }
  atomicAdd(&ssum, lsum);
  __syncthreads();
  if (tid == 0) srms = rsqrtf(ssum * (1.0f / (float)K_DIM) + RMS_EPS);
  __syncthreads();
  const float rms = srms;

  if (tid < 8 * M_COEF) {
    const int j  = tid % M_COEF;
    const int sl = tid / M_COEF;
    const int k0 = sl * 128;
    float acc = 0.0f;
    for (int k = 0; k < 128; ++k)
      acc += xf[k0 + k] * bf16_to_f32(phi[(long)(k0 + k) * M_COEF + j]);
    part[tid] = acc * rms;
  }
  __syncthreads();
  if (tid < M_COEF) {
    float acc = bvec[tid];
    for (int s = 0; s < 8; ++s) acc += part[s * M_COEF + tid];
    logits[tid] = acc;
  }
  __syncthreads();

  if (tid == 0) {
    const float ap = a_pre[0], apo = a_post[0], ar = a_res[0];
    float M[HC * HC];
#pragma unroll
    for (int h = 0; h < HC; ++h) {
      hpre[h]  = sigmoidf(ap * logits[h]) + PRE_EPS;
      hpost[h] = POST_MULT * sigmoidf(apo * logits[HC + h]);
    }
#pragma unroll
    for (int i = 0; i < HC * HC; ++i) M[i] = __expf(ar * logits[2 * HC + i]);
    for (int t = 0; t < TMAX_IT; ++t) {
#pragma unroll
      for (int r = 0; r < HC; ++r) {
        float s = SINK_EPS;
        for (int c = 0; c < HC; ++c) s += M[r * HC + c];
        const float inv = 1.0f / s;
        for (int c = 0; c < HC; ++c) M[r * HC + c] *= inv;
      }
#pragma unroll
      for (int c = 0; c < HC; ++c) {
        float s = SINK_EPS;
        for (int r = 0; r < HC; ++r) s += M[r * HC + c];
        const float inv = 1.0f / s;
        for (int r = 0; r < HC; ++r) M[r * HC + c] *= inv;
      }
    }
#pragma unroll
    for (int i = 0; i < HC * HC; ++i) hres[i] = M[i];
  }
  __syncthreads();

  const int c = tid;
  float xi = 0.0f;
#pragma unroll
  for (int h = 0; h < HC; ++h) xi += xf[h * C_DIM + c] * hpre[h];
  xin[c] = xi;
  __syncthreads();

  float fo = 0.0f;
  const float* wrow = w_inner + (long)c * C_DIM;
  for (int i = 0; i < C_DIM; i += 4) {
    const float4 w = *(const float4*)(wrow + i);
    fo += xin[i] * w.x + xin[i + 1] * w.y + xin[i + 2] * w.z + xin[i + 3] * w.w;
  }

#pragma unroll
  for (int h = 0; h < HC; ++h) {
    float v = hpost[h] * fo;
#pragma unroll
    for (int h2 = 0; h2 < HC; ++h2) v += hres[h * HC + h2] * xf[h2 * C_DIM + c];
    xm[(long)tok * K_DIM + h * C_DIM + c] = f32_to_bf16(v);
  }
}

// -------------------- Kernel 2: head GEMM via WMMA bf16 + TDM --------------------
#define TM 128
#define TN 128
#define TK 32

__device__ inline v16bf ld_frag2(const unsigned short* p0, const unsigned short* p1) {
  union { uint4 u[2]; v16bf v; } r;
  r.u[0] = *(const uint4*)p0;
  r.u[1] = *(const uint4*)p1;
  return r.v;
}

__global__ __launch_bounds__(256) void head_gemm_kernel(
    const unsigned short* __restrict__ A,    // [NTOK][1024] bf16
    const float*          __restrict__ Bw,   // [VOCAB][1024] f32
    float*                __restrict__ out)  // [NTOK][VOCAB]
{
  __shared__ __align__(16) unsigned short lA[TM * TK];  // 8 KB
  __shared__ __align__(16) unsigned short lB[TN * TK];  // 8 KB

  const int tid  = threadIdx.x;
  const int lane = tid & 31;
  const int wave = tid >> 5;     // 0..7
  const int wm   = wave >> 1;    // 0..3 -> 32-row M strip
  const int wn   = wave & 1;     // 0..1 -> 64-col N strip
  // m-tiles on blockIdx.x (fast) so consecutive WGs reuse the same w_head slice in L2
  const int m0   = blockIdx.x * TM;
  const int n0   = blockIdx.y * TN;

  const int lrow  = tid >> 1;          // 0..127
  const int lhalf = (tid & 1) * 16;    // 0 or 16 (K elements)

  const v8f vzero = {0.f, 0.f, 0.f, 0.f, 0.f, 0.f, 0.f, 0.f};
  v8f acc[2][4];
#pragma unroll
  for (int i = 0; i < 2; ++i)
#pragma unroll
    for (int j = 0; j < 4; ++j) acc[i][j] = vzero;

  int nrow = n0 + lrow;
  if (nrow >= VOCAB) nrow = VOCAB - 1;   // clamp; stores are guarded
  const float* bptr = Bw + (size_t)nrow * K_DIM + lhalf;

#if __has_builtin(__builtin_amdgcn_tensor_load_to_lds) && __has_builtin(__builtin_amdgcn_s_wait_tensorcnt)
#define USE_TDM 1
  const unsigned lds_lA = (unsigned)(uintptr_t)&lA[0];  // flat LDS addr: offset in low 32 bits
#else
#define USE_TDM 0
  const unsigned short* aptr = A + (size_t)(m0 + lrow) * K_DIM + lhalf;
#endif

  for (int k0 = 0; k0 < K_DIM; k0 += TK) {
#if USE_TDM
    // ---- A tile via Tensor Data Mover: 32(K) x 128(M) tile of the 1024x4096 bf16 tensor ----
    if (wave == 0) {
      const unsigned long long gaddr =
          (unsigned long long)(uintptr_t)A + ((size_t)m0 * K_DIM + (size_t)k0) * 2ull;
      u32x4 g0;
      g0[0] = 1u;                                        // count=1 (valid user D#)
      g0[1] = lds_lA;                                    // lds_addr
      g0[2] = (unsigned)(gaddr & 0xFFFFFFFFull);         // global_addr[31:0]
      g0[3] = (unsigned)((gaddr >> 32) & 0x1FFFFFFull)   // global_addr[56:32]
              | (2u << 30);                              // type=2 ("image")
      i32x8 g1;
      g1[0] = 0x10000;                                   // data_size=1 (2 bytes/elem)
      g1[1] = (int)((K_DIM & 0xFFFF) << 16);             // tensor_dim0[15:0]
      g1[2] = (int)(((K_DIM >> 16) & 0xFFFF)             // tensor_dim0[31:16]
              | ((NTOK & 0xFFFF) << 16));                // tensor_dim1[15:0]
      g1[3] = (int)(((NTOK >> 16) & 0xFFFF)              // tensor_dim1[31:16]
              | (TK << 16));                             // tile_dim0 = 32 (K)
      g1[4] = TM;                                        // tile_dim1 = 128 (M rows), tile_dim2 = 0
      g1[5] = K_DIM;                                     // tensor_dim0_stride[31:0]
      g1[6] = 0;                                         // stride hi / dim1_stride (2D: unused)
      g1[7] = 0;
      const i32x4 gz4 = {0, 0, 0, 0};
      const i32x8 gz8 = {0, 0, 0, 0, 0, 0, 0, 0};
      __builtin_amdgcn_tensor_load_to_lds(g0, g1, gz4, gz4, gz8, 0);
    }
#else
    const uint4 a0 = *(const uint4*)(aptr);
    const uint4 a1 = *(const uint4*)(aptr + 8);
    *(uint4*)&lA[lrow * TK + lhalf]     = a0;
    *(uint4*)&lA[lrow * TK + lhalf + 8] = a1;
    aptr += TK;
#endif

    // ---- stage B tile (w_head f32 -> bf16) overlapped with the TDM transfer ----
    __align__(16) float fb[16];
    *(float4*)&fb[0]  = *(const float4*)(bptr);
    *(float4*)&fb[4]  = *(const float4*)(bptr + 4);
    *(float4*)&fb[8]  = *(const float4*)(bptr + 8);
    *(float4*)&fb[12] = *(const float4*)(bptr + 12);
    __align__(16) unsigned hb[8];
#pragma unroll
    for (int i = 0; i < 8; ++i) hb[i] = pack_bf16x2(fb[2 * i], fb[2 * i + 1]);
    *(uint4*)&lB[lrow * TK + lhalf]     = *(const uint4*)&hb[0];
    *(uint4*)&lB[lrow * TK + lhalf + 8] = *(const uint4*)&hb[4];

    if (k0 + TK < K_DIM)
      __builtin_prefetch((const void*)(bptr + TK), 0, 1);

#if USE_TDM
    if (wave == 0) __builtin_amdgcn_s_wait_tensorcnt(0);  // A tile landed in LDS
#endif
    __syncthreads();

    // ---- fragments per 16-bit WMMA VGPR layouts ----
    const int aksel = (lane >> 4) * 8;    // A: lanes 0-15 -> K{0..7,16..23}; 16-31 -> K{8..15,24..31}
    const int bksel = (lane >> 4) * 16;   // unused: B loads use aksel, same K order as A
    v16bf afrag[2], bfrag[4];
#pragma unroll
    for (int i = 0; i < 2; ++i) {
      const int m = wm * 32 + i * 16 + (lane & 15);
      afrag[i] = ld_frag2(&lA[m * TK + aksel], &lA[m * TK + 16 + aksel]);
    }
#pragma unroll
    for (int j = 0; j < 4; ++j) {
      const int n = wn * 64 + j * 16 + (lane & 15);
      bfrag[j] = ld_frag2(&lB[n * TK + aksel], &lB[n * TK + 16 + aksel]);
    }

#pragma unroll
    for (int i = 0; i < 2; ++i)
#pragma unroll
      for (int j = 0; j < 4; ++j)
        acc[i][j] = __builtin_amdgcn_wmma_f32_16x16x32_bf16(
            false, afrag[i], false, bfrag[j], (short)0, acc[i][j], false, false);

    __syncthreads();
    bptr += TK;
  }

  // ---- store C fragments: VGPR r -> M = r (+8 for lanes 16..31), N = lane&15 ----
#pragma unroll
  for (int i = 0; i < 2; ++i) {
    const int mrow = m0 + wm * 32 + i * 16 + ((lane >> 4) << 3);
#pragma unroll
    for (int j = 0; j < 4; ++j) {
      const int n = n0 + wn * 64 + j * 16 + (lane & 15);
      if (n < VOCAB) {
        float* op = out + (size_t)mrow * VOCAB + n;
#pragma unroll
        for (int r = 0; r < 8; ++r) op[(size_t)r * VOCAB] = acc[i][j][r];
      }
    }
  }
}

// -------------------- launcher --------------------
extern "C" void kernel_launch(void* const* d_in, const int* in_sizes, int n_in,
                              void* d_out, int out_size, void* d_ws, size_t ws_size,
                              hipStream_t stream) {
  (void)in_sizes; (void)n_in; (void)out_size; (void)ws_size;
  const int*            ids     = (const int*)d_in[0];
  const float*          embed   = (const float*)d_in[1];
  const float*          w_inner = (const float*)d_in[2];
  const float*          w_head  = (const float*)d_in[3];
  const unsigned short* phi     = (const unsigned short*)d_in[4];   // bf16 bits
  const float*          bvec    = (const float*)d_in[5];
  const float*          a_pre   = (const float*)d_in[6];
  const float*          a_post  = (const float*)d_in[7];
  const float*          a_res   = (const float*)d_in[8];

  unsigned short* xm  = (unsigned short*)d_ws;  // 4096*1024 bf16 = 8 MB
  float*          out = (float*)d_out;

  mhc_token_kernel<<<NTOK, 256, 0, stream>>>(ids, embed, w_inner, phi, bvec,
                                             a_pre, a_post, a_res, xm);

  // m-tiles fast-varying (blockIdx.x) for w_head L2 reuse
  dim3 grid(NTOK / TM, (VOCAB + TN - 1) / TN);  // 32 x 393
  head_gemm_kernel<<<grid, 256, 0, stream>>>(xm, w_head, out);
}
